// UnifiedMTNet_61744449847597
// MI455X (gfx1250) — matrix operand order repaired and checked by reading, not compiled, not run on hardware
//
#include <hip/hip_runtime.h>
#include <hip/hip_bf16.h>

// ---------------- problem constants ----------------
#define JNT    24
#define BATCH  8192
#define NNODE  (BATCH * JNT)      // 196608
#define NCHAIN 46                 // 2*(J-1) chain edges per batch
#define BT     8                  // batches per workgroup in spatial kernels

typedef __attribute__((ext_vector_type(16))) _Float16 v16h;
typedef __attribute__((ext_vector_type(8)))  float    v8f;

// fast transcendentals: v_exp_f32 / v_log_f32 / v_rcp_f32 / v_sqrt_f32
__device__ __forceinline__ float sigmoidf_(float x) {
  return __builtin_amdgcn_rcpf(1.f + __expf(-x));
}
__device__ __forceinline__ float softplusf_(float x) {
  return fmaxf(x, 0.f) + __logf(1.f + __expf(-fabsf(x)));
}
__device__ __forceinline__ float frcp_(float x)  { return __builtin_amdgcn_rcpf(x); }
__device__ __forceinline__ float fsqrt_(float x) { return __builtin_amdgcn_sqrtf(x); }
__device__ __forceinline__ v8f   v8f_zero() { v8f z = {0.f,0.f,0.f,0.f,0.f,0.f,0.f,0.f}; return z; }
__device__ __forceinline__ int   imin_(int a, int b) { return a < b ? a : b; }
__device__ __forceinline__ int   imax_(int a, int b) { return a > b ? a : b; }

// ---------------- WMMA fragment loaders ----------------
// A (16x32 f16): lanes 0-15 row M=lane hold K {kofs..kofs+7, kofs+16..kofs+23} with kofs=0,
// lanes 16-31 row M=lane-16 with kofs=8.  (ISA 7.12.2)
__device__ __forceinline__ v16h load_afrag(const _Float16* st, int kpad, int kt, int lane) {
  const int row  = lane & 15;
  const int kofs = (lane & 16) ? 8 : 0;
  const _Float16* p = st + row * kpad + kt * 32 + kofs;
  v16h a;
#pragma unroll
  for (int i = 0; i < 8; ++i) { a[i] = p[i]; a[8 + i] = p[16 + i]; }
  return a;
}

// B (32x16 f16): element i of the v16h is K = (lane>=16?8:0)+i+(i>=8?8:0), column N = lane&15.
// B[k][n] = W[n][k]   (W is row-major OUT x KW)
template<int OUT>
__device__ __forceinline__ v16h load_bfrag(const float* __restrict__ W, int kw,
                                           int kt, int nt, int lane) {
  const int n     = nt * 16 + (lane & 15);
  const int kbase = kt * 32 + ((lane & 16) ? 8 : 0);
  v16h b;
#pragma unroll
  for (int i = 0; i < 16; ++i) {
    const int k = kbase + i + ((i >= 8) ? 8 : 0);
    const float v = (n < OUT && k < kw) ? W[n * kw + k] : 0.f;
    b[i] = (_Float16)v;
  }
  return b;
}

// ---------------- feature accessor (node-major tile in LDS) ----------------
struct LdsFeat {
  const float* p; int in;
  __device__ float operator()(int bb, int j, int k) const {
    return p[(bb * JNT + j) * in + k];
  }
};

// ---------------- one spatial block (messages via WMMA, agg via LDS float atomics) ---
template<int IN, int OUT, int KPAD, class F>
__device__ void run_block(const F& feat, const float* __restrict__ ea, long b0,
                          const float* __restrict__ Wf, const float* __restrict__ Ws,
                          const float* __restrict__ Wu,
                          const float* __restrict__ bf, const float* __restrict__ bs,
                          const float* __restrict__ bu,
                          float* agg /* LDS [BT*J*OUT] */,
                          _Float16* stageAll, int stageStride) {
  constexpr int KT = KPAD / 32;
  constexpr int NT = (OUT + 15) / 16;
  constexpr int KW = 2 * IN + 3;
  const int tid  = threadIdx.x;
  const int lane = tid & 31;
  const int wave = tid >> 5;

  // 1) self/update path:  agg = bu + Wu * x[node]
  for (int node = tid; node < BT * JNT; node += 256) {
    const int bb = node / JNT, j = node - bb * JNT;
    float fv[IN];
#pragma unroll
    for (int k = 0; k < IN; ++k) fv[k] = feat(bb, j, k);
    for (int ch = 0; ch < OUT; ++ch) {
      float acc = bu[ch];
#pragma unroll
      for (int k = 0; k < IN; ++k) acc = fmaf(Wu[ch * IN + k], fv[k], acc);
      agg[node * OUT + ch] = acc;
    }
  }

  // 2) loop-invariant B fragments + gate biases in registers
  v16h Bfr[KT][NT], Bsr[KT][NT];
#pragma unroll
  for (int kt = 0; kt < KT; ++kt)
#pragma unroll
    for (int nt = 0; nt < NT; ++nt) {
      Bfr[kt][nt] = load_bfrag<OUT>(Wf, KW, kt, nt, lane);
      Bsr[kt][nt] = load_bfrag<OUT>(Ws, KW, kt, nt, lane);
    }
  float bfv[NT], bsv[NT];
#pragma unroll
  for (int nt = 0; nt < NT; ++nt) {
    const int n = nt * 16 + (lane & 15);
    bfv[nt] = (n < OUT) ? bf[n] : 0.f;
    bsv[nt] = (n < OUT) ? bs[n] : 0.f;
  }
  __syncthreads();

  // 3) message tiles: 23 tiles of (2 chain edges x 8 batches) = 16 WMMA rows
  _Float16* st = stageAll + wave * stageStride;
  for (int t = wave; t < NCHAIN / 2; t += 8) {
    // build z rows in per-wave f16 staging: z = [x[dst], x[src], edge_attr]
    // four contiguous range-loops (no per-element branching)
    {
      const int row = lane & 15;
      const int c   = 2 * t + (row >> 3);
      const int bb  = row & 7;
      const int jd  = (c < JNT - 1) ? c + 1 : c - (JNT - 1);
      const int js  = (c < JNT - 1) ? c     : c - (JNT - 2);
      const int k0  = (lane >> 4) * (KPAD / 2);
      const int k1  = k0 + (KPAD / 2);
      _Float16* sr  = st + row * KPAD;
      const float* eap = ea + ((long)c * BATCH + b0 + bb) * 3;
      for (int k = k0;                 k < imin_(k1, IN);     ++k) sr[k] = (_Float16)feat(bb, jd, k);
      for (int k = imax_(k0, IN);      k < imin_(k1, 2 * IN); ++k) sr[k] = (_Float16)feat(bb, js, k - IN);
      for (int k = imax_(k0, 2 * IN);  k < imin_(k1, KW);     ++k) sr[k] = (_Float16)eap[k - 2 * IN];
      for (int k = imax_(k0, KW);      k < k1;                ++k) sr[k] = (_Float16)0.f;
    }
    // WMMA gate GEMMs:  F = z*Wf^T,  S = z*Ws^T  (f32 accumulate)
    v8f accF[NT], accS[NT];
#pragma unroll
    for (int nt = 0; nt < NT; ++nt) { accF[nt] = v8f_zero(); accS[nt] = v8f_zero(); }
#pragma unroll
    for (int kt = 0; kt < KT; ++kt) {
      v16h A = load_afrag(st, KPAD, kt, lane);
#pragma unroll
      for (int nt = 0; nt < NT; ++nt) {
        accF[nt] = __builtin_amdgcn_wmma_f32_16x16x32_f16(false, A, false, Bfr[kt][nt],
                                                          (short)0, accF[nt], false, false);
        accS[nt] = __builtin_amdgcn_wmma_f32_16x16x32_f16(false, A, false, Bsr[kt][nt],
                                                          (short)0, accS[nt], false, false);
      }
    }
    // nonlinearity + scatter into node accumulators (ds_add_f32)
    const int nloc  = lane & 15;
    const int mbase = (lane >> 4) * 8;   // C layout: VGPR r -> M = mbase+r, N = nloc
#pragma unroll
    for (int nt = 0; nt < NT; ++nt) {
      const int n = nt * 16 + nloc;
      if (n < OUT) {
#pragma unroll
        for (int r = 0; r < 8; ++r) {
          const int m  = mbase + r;
          const int cc = 2 * t + (m >> 3);
          const int mb = m & 7;
          const int dj = (cc < JNT - 1) ? cc + 1 : cc - (JNT - 1);
          const float mg = sigmoidf_(accF[nt][r] + bfv[nt]) * softplusf_(accS[nt][r] + bsv[nt]);
          atomicAdd(&agg[(mb * JNT + dj) * OUT + n], mg);
        }
      }
    }
  }
  __syncthreads();
}

// ---------------- fused two-spatial-block kernel ----------------
// LDS plan: mid1 | uni (x-tile UNION block-2 output) | per-wave f16 staging
// x-tile is filled via global_load_async_to_lds_b128 (ASYNCcnt) and is dead after
// block 1's message phase; block 2 reuses the same LDS for its output accumulator.
template<int IN1, int OUT1, int OUT2, int K1PAD, int K2PAD>
__global__ __launch_bounds__(256)
void fused_sb2_kernel(const float* __restrict__ xin, const float* __restrict__ ea,
                      const float* Wf1, const float* bf1, const float* Ws1,
                      const float* bs1, const float* Wu1, const float* bu1,
                      const float* Wf2, const float* bf2, const float* Ws2,
                      const float* bs2, const float* Wu2, const float* bu2,
                      float* __restrict__ out) {
  constexpr int KSMAX = (K1PAD > K2PAD) ? K1PAD : K2PAD;
  constexpr int UNIF  = (BT * JNT * IN1 > BT * JNT * OUT2) ? BT * JNT * IN1 : BT * JNT * OUT2;
  __shared__ float     mid1[BT * JNT * OUT1];
  __shared__ float     uni[UNIF];            // x-tile, later reused as out2
  __shared__ _Float16  stage[8 * 16 * KSMAX];
  const long b0 = (long)blockIdx.x * BT;

  // ---- async copy of the input tile (BT*J*IN1 floats, contiguous) into LDS ----
  {
    const float* gsrc = xin + b0 * JNT * IN1;          // uniform (SGPR) base
    const unsigned lds_base = (unsigned)(unsigned long long)(void*)uni;
    constexpr int CHUNKS = (BT * JNT * IN1) / 4;       // 16-byte chunks
    for (int c = threadIdx.x; c < CHUNKS; c += 256) {
      const unsigned laddr = lds_base + (unsigned)c * 16u;
      const unsigned goff  = (unsigned)c * 16u;
      asm volatile("global_load_async_to_lds_b128 %0, %1, %2"
                   :: "v"(laddr), "v"(goff), "s"(gsrc) : "memory");
    }
    asm volatile("s_wait_asynccnt 0" ::: "memory");
  }
  __syncthreads();

  LdsFeat f1{uni, IN1};
  run_block<IN1, OUT1, K1PAD>(f1, ea, b0, Wf1, Ws1, Wu1, bf1, bs1, bu1,
                              mid1, stage, 16 * KSMAX);
  // x-tile dead beyond this point: reuse 'uni' as block-2 output accumulator
  LdsFeat f2{mid1, OUT1};
  run_block<OUT1, OUT2, K2PAD>(f2, ea, b0, Wf2, Ws2, Wu2, bf2, bs2, bu2,
                               uni, stage, 16 * KSMAX);

  for (int idx = threadIdx.x; idx < BT * JNT * OUT2; idx += 256)
    out[b0 * JNT * OUT2 + idx] = uni[idx];
}

// ---------------- causal TCN (edge-padded, kernel 3) + optional denorm -------------
template<int C>
__global__ __launch_bounds__(256)
void tcn_kernel(const float* __restrict__ hin, const float* __restrict__ w,
                const float* __restrict__ bias, const float* __restrict__ scale,
                const float* __restrict__ shift, float* __restrict__ out) {
  __shared__ float wl[C * C * 3];
  __shared__ float bl[C];
  for (int i = threadIdx.x; i < C * C * 3; i += 256) wl[i] = w[i];
  for (int i = threadIdx.x; i < C; i += 256) bl[i] = bias[i];
  __syncthreads();

  const long gid = (long)blockIdx.x * 256 + threadIdx.x;  // over NNODE*C exactly
  const int  o   = (int)(gid % C);
  const long n   = gid / C;
  const int  b   = (int)(n / JNT);
  const int  j   = (int)(n - (long)b * JNT);

  float acc = bl[o];
#pragma unroll
  for (int k = 0; k < 3; ++k) {
    int bs = b + k - 2; if (bs < 0) bs = 0;            // 'edge' pad (replicate frame 0)
    const float* hrow = hin + ((long)bs * JNT + j) * C;
    __builtin_prefetch(hrow, 0, 3);
    for (int i = 0; i < C; ++i) acc = fmaf(wl[(o * C + i) * 3 + k], hrow[i], acc);
  }
  float y = fmaxf(acc, 0.f) + hin[n * C + o];           // relu(y+b) + x
  if (scale) y = y * scale[n * C + o] + shift[n * C + o];
  out[n * C + o] = y;
}

// ---------------- root path (only joint 0 of root1/root2 is ever used) -------------
__global__ __launch_bounds__(256)
void root_kernel(const float* __restrict__ x, const float* __restrict__ ea,
                 const float* __restrict__ eat, const float* __restrict__ node_attr,
                 const float* W1f, const float* b1f, const float* W1s, const float* b1s,
                 const float* W1u, const float* b1u,
                 const float* W2f, const float* b2f, const float* W2s, const float* b2s,
                 const float* W2u, const float* b2u,
                 const float* __restrict__ root_mean, const float* __restrict__ root_std,
                 float* __restrict__ out_root) {
  const int b = blockIdx.x * 256 + threadIdx.x;
  if (b >= BATCH) return;
  const long e = (long)(JNT - 1) * BATCH + b;  // chain edge c=23: src=(b,1) -> dst=(b,0)

  float z9[9];
#pragma unroll
  for (int i = 0; i < 3; ++i) {
    z9[i]     = x[((long)b * JNT + 0) * 9 + 6 + i];
    z9[3 + i] = x[((long)b * JNT + 1) * 9 + 6 + i];
    z9[6 + i] = ea[e * 3 + i];
  }
  float r0[16];
  for (int ch = 0; ch < 16; ++ch) {
    float f = b1f[ch], s = b1s[ch];
#pragma unroll
    for (int k = 0; k < 9; ++k) {
      f = fmaf(W1f[ch * 9 + k], z9[k], f);
      s = fmaf(W1s[ch * 9 + k], z9[k], s);
    }
    float u = b1u[ch];
#pragma unroll
    for (int k = 0; k < 3; ++k) u = fmaf(W1u[ch * 3 + k], z9[k], u);
    r0[ch] = sigmoidf_(f) * softplusf_(s) + u;
  }
  float z41[41];
#pragma unroll
  for (int ch = 0; ch < 16; ++ch) { z41[ch] = r0[ch]; z41[19 + ch] = r0[ch]; }
#pragma unroll
  for (int i = 0; i < 3; ++i) {
    z41[16 + i] = node_attr[((long)b * JNT + 0) * 3 + i];
    z41[35 + i] = node_attr[((long)b * JNT + 1) * 3 + i];
    z41[38 + i] = eat[e * 3 + i];
  }
  for (int ch = 0; ch < 3; ++ch) {
    float f = b2f[ch], s = b2s[ch];
    for (int k = 0; k < 41; ++k) {
      f = fmaf(W2f[ch * 41 + k], z41[k], f);
      s = fmaf(W2s[ch * 41 + k], z41[k], s);
    }
    float u = b2u[ch];
    for (int k = 0; k < 19; ++k) u = fmaf(W2u[ch * 19 + k], z41[k], u);
    const float rp = sigmoidf_(f) * softplusf_(s) + u;
    out_root[b * 3 + ch] = rp * root_std[b * 3 + ch] + root_mean[b * 3 + ch];
  }
}

// ---------------- FK chain + final pos / fake_x assembly ----------------
__global__ __launch_bounds__(256)
void fk_kernel(const float* __restrict__ ang, const float* __restrict__ offset,
               const float* __restrict__ root, const float* __restrict__ mean,
               const float* __restrict__ std_, float* __restrict__ pos_out,
               float* __restrict__ fake_out) {
  const int b = blockIdx.x * 256 + threadIdx.x;
  if (b >= BATCH) return;
  float g[9];
  float p[3] = {0.f, 0.f, 0.f};
  float rp[3];
#pragma unroll
  for (int i = 0; i < 3; ++i) rp[i] = root[b * 3 + i];

  for (int j = 0; j < JNT; ++j) {
    const long n = (long)b * JNT + j;
    float d6[6];
#pragma unroll
    for (int i = 0; i < 6; ++i) d6[i] = ang[n * 6 + i];
    // rot6d -> R  (columns b1,b2,b3); fast v_sqrt / v_rcp
    float i1 = frcp_(fsqrt_(d6[0]*d6[0] + d6[1]*d6[1] + d6[2]*d6[2]) + 1e-8f);
    float b1[3] = {d6[0]*i1, d6[1]*i1, d6[2]*i1};
    float dp = b1[0]*d6[3] + b1[1]*d6[4] + b1[2]*d6[5];
    float b2[3] = {d6[3]-dp*b1[0], d6[4]-dp*b1[1], d6[5]-dp*b1[2]};
    float i2 = frcp_(fsqrt_(b2[0]*b2[0] + b2[1]*b2[1] + b2[2]*b2[2]) + 1e-8f);
    b2[0]*=i2; b2[1]*=i2; b2[2]*=i2;
    float b3[3] = {b1[1]*b2[2]-b1[2]*b2[1], b1[2]*b2[0]-b1[0]*b2[2], b1[0]*b2[1]-b1[1]*b2[0]};
    float R[9] = {b1[0],b2[0],b3[0], b1[1],b2[1],b3[1], b1[2],b2[2],b3[2]};

    if (j == 0) {
#pragma unroll
      for (int i = 0; i < 9; ++i) g[i] = R[i];
      p[0] = p[1] = p[2] = 0.f;
    } else {
      float off[3];
#pragma unroll
      for (int i = 0; i < 3; ++i) off[i] = offset[n * 3 + i];
      float np[3], ng[9];
#pragma unroll
      for (int r = 0; r < 3; ++r) {
        np[r] = p[r] + g[r*3+0]*off[0] + g[r*3+1]*off[1] + g[r*3+2]*off[2];
#pragma unroll
        for (int c = 0; c < 3; ++c)
          ng[r*3+c] = g[r*3+0]*R[0*3+c] + g[r*3+1]*R[1*3+c] + g[r*3+2]*R[2*3+c];
      }
#pragma unroll
      for (int i = 0; i < 9; ++i) g[i] = ng[i];
      p[0] = np[0]; p[1] = np[1]; p[2] = np[2];
    }
    const float pw[3] = {p[0]+rp[0], p[1]+rp[1], p[2]+rp[2]};
#pragma unroll
    for (int i = 0; i < 3; ++i) pos_out[n * 3 + i] = pw[i];
#pragma unroll
    for (int q = 0; q < 6; ++q)
      fake_out[n * 9 + q] = (d6[q] - mean[n * 9 + q]) * frcp_(std_[n * 9 + q]);
#pragma unroll
    for (int q = 0; q < 3; ++q)
      fake_out[n * 9 + 6 + q] = (pw[q] - mean[n * 9 + 6 + q]) * frcp_(std_[n * 9 + 6 + q]);
  }
}

// ---------------- launcher ----------------
#define PF(i) ((const float*)d_in[(i)])

extern "C" void kernel_launch(void* const* d_in, const int* in_sizes, int n_in,
                              void* d_out, int out_size, void* d_ws, size_t ws_size,
                              hipStream_t stream) {
  (void)in_sizes; (void)n_in; (void)out_size; (void)ws_size;

  const float* x         = PF(0);
  const float* ea        = PF(2);
  const float* eat       = PF(3);
  const float* node_attr = PF(4);
  const float* offset    = PF(6);
  const float* ang_mean  = PF(7);
  const float* ang_std   = PF(8);
  const float* root_mean = PF(9);
  const float* root_std  = PF(10);
  const float* mean      = PF(11);
  const float* stdv      = PF(12);

  // params flattened in setup_inputs() insertion order:
  // enc1{Wf,bf,Ws,bs,Wu,bu}, enc2{...}, enc_w, enc_b,
  // dec1{...}, dec2{...}, dec_w, dec_b, root1{...}, root2{...}
  const float *e1Wf = PF(13), *e1bf = PF(14), *e1Ws = PF(15), *e1bs = PF(16), *e1Wu = PF(17), *e1bu = PF(18);
  const float *e2Wf = PF(19), *e2bf = PF(20), *e2Ws = PF(21), *e2bs = PF(22), *e2Wu = PF(23), *e2bu = PF(24);
  const float *enc_w = PF(25), *enc_b = PF(26);
  const float *d1Wf = PF(27), *d1bf = PF(28), *d1Ws = PF(29), *d1bs = PF(30), *d1Wu = PF(31), *d1bu = PF(32);
  const float *d2Wf = PF(33), *d2bf = PF(34), *d2Ws = PF(35), *d2bs = PF(36), *d2Wu = PF(37), *d2bu = PF(38);
  const float *dec_w = PF(39), *dec_b = PF(40);
  const float *r1Wf = PF(41), *r1bf = PF(42), *r1Ws = PF(43), *r1bs = PF(44), *r1Wu = PF(45), *r1bu = PF(46);
  const float *r2Wf = PF(47), *r2bf = PF(48), *r2Ws = PF(49), *r2bs = PF(50), *r2Wu = PF(51), *r2bu = PF(52);

  // workspace: h (N*32), z (N*32), d (N*6)
  float* ws_h = (float*)d_ws;
  float* ws_z = ws_h + (size_t)NNODE * 32;
  float* ws_d = ws_z + (size_t)NNODE * 32;

  // output layout: ang (B*J*6) | pos (B*J*3) | root (B*3) | fake_x (N*9)
  float* ang   = (float*)d_out;
  float* pos   = ang + (size_t)BATCH * JNT * 6;
  float* rootp = pos + (size_t)BATCH * JNT * 3;
  float* fake  = rootp + (size_t)BATCH * 3;

  // 1) encoder spatial blocks (9 -> 16 -> 32), edge_attr
  fused_sb2_kernel<9, 16, 32, 32, 64><<<BATCH / BT, 256, 0, stream>>>(
      x, ea, e1Wf, e1bf, e1Ws, e1bs, e1Wu, e1bu,
             e2Wf, e2bf, e2Ws, e2bs, e2Wu, e2bu, ws_h);

  // 2) encoder TCN over frames
  tcn_kernel<32><<<(NNODE * 32) / 256, 256, 0, stream>>>(
      ws_h, enc_w, enc_b, nullptr, nullptr, ws_z);

  // 3) decoder spatial blocks (32 -> 16 -> 6), edge_attr_t
  fused_sb2_kernel<32, 16, 6, 96, 64><<<BATCH / BT, 256, 0, stream>>>(
      ws_z, eat, d1Wf, d1bf, d1Ws, d1bs, d1Wu, d1bu,
              d2Wf, d2bf, d2Ws, d2bs, d2Wu, d2bu, ws_d);

  // 4) decoder TCN fused with ang denormalization -> d_out[ang]
  tcn_kernel<6><<<(NNODE * 6) / 256, 256, 0, stream>>>(
      ws_d, dec_w, dec_b, ang_std, ang_mean, ang);

  // 5) root position -> d_out[root]
  root_kernel<<<BATCH / 256, 256, 0, stream>>>(
      x, ea, eat, node_attr,
      r1Wf, r1bf, r1Ws, r1bs, r1Wu, r1bu,
      r2Wf, r2bf, r2Ws, r2bs, r2Wu, r2bu,
      root_mean, root_std, rootp);

  // 6) FK + pos + fake_x
  fk_kernel<<<BATCH / 256, 256, 0, stream>>>(
      ang, offset, rootp, mean, stdv, pos, fake);
}
